// LLaDAExpertGroup_24936580120994
// MI455X (gfx1250) — matrix operand order repaired and checked by reading, not compiled, not run on hardware
//
#include <hip/hip_runtime.h>
#include <hip/hip_bf16.h>

// ---------------------------------------------------------------------------
// Types / helpers
// ---------------------------------------------------------------------------
typedef __bf16 bf16;
typedef __attribute__((ext_vector_type(16))) __bf16 v16bf;
typedef __attribute__((ext_vector_type(8)))  __bf16 v8bf;
typedef __attribute__((ext_vector_type(8)))  float  v8f;

__device__ __forceinline__ float bf2f(bf16 h) {
  union { unsigned short s; bf16 b; } u; u.b = h;
  union { unsigned int i; float f; } v; v.i = ((unsigned int)u.s) << 16;
  return v.f;
}
__device__ __forceinline__ bf16 f2bf(float f) {
  union { unsigned int i; float f; } v; v.f = f;
  unsigned int r = v.i + 0x7fffu + ((v.i >> 16) & 1u);   // RNE
  union { unsigned short s; bf16 b; } u; u.s = (unsigned short)(r >> 16);
  return u.b;
}
__device__ __forceinline__ v16bf ld16(const bf16* lo, const bf16* hi) {
  v8bf a = *(const v8bf*)lo;
  v8bf b = *(const v8bf*)hi;
  return __builtin_shufflevector(a, b, 0,1,2,3,4,5,6,7,8,9,10,11,12,13,14,15);
}
__device__ __forceinline__ v8f vzero8() { v8f z = {}; return z; }
__device__ __forceinline__ float silu_clip(float v) {
  v = fminf(5.0f, fmaxf(-5.0f, v));
  return v / (1.0f + __expf(-v));
}

// Problem constants (match reference)
enum { Bc = 4, Sc = 4096, Dc = 1024, Hc = 2048, Ac = 128, Ec = 8, Mc = Bc * Sc };

// ---------------------------------------------------------------------------
// Tensor Data Mover support (CDNA5).  D# bitfields per CDNA5 ISA.
// ---------------------------------------------------------------------------
#if __has_builtin(__builtin_amdgcn_tensor_load_to_lds) && \
    __has_builtin(__builtin_amdgcn_s_wait_tensorcnt)
#define HAVE_TDM 1
typedef unsigned int u32x4 __attribute__((ext_vector_type(4)));
typedef int          i32x4 __attribute__((ext_vector_type(4)));
typedef int          i32x8 __attribute__((ext_vector_type(8)));

// 2-D bf16 tile load: tile_k x tile_rows from a row-major [rows x K] tensor,
// packed into LDS with 4 DWORDs of padding every 16 DWORDs (-> 40-elem rows).
__device__ __forceinline__ void tdm_load_2d(unsigned lds_off, const bf16* gsrc,
                                            unsigned tile_k, unsigned tile_rows,
                                            unsigned K, unsigned rows) {
  const unsigned long long ga = (unsigned long long)(size_t)gsrc;
  u32x4 g0 = {};
  g0[0] = 1u;                                          // count=1, user mode
  g0[1] = lds_off;                                     // lds_addr
  g0[2] = (unsigned)(ga & 0xffffffffu);                // global_addr[31:0]
  g0[3] = (unsigned)((ga >> 32) & 0x1ffffffu)          // global_addr[56:32]
        | (2u << 30);                                  // type=2 ("image")
  i32x8 g1 = {};
  g1[0] = (int)((1u << 16)                             // data_size = 2B
        |       (1u << 20)                             // pad_enable
        |       (3u << 22)                             // pad_interval: 16 DWORDs
        |       (3u << 25));                           // pad_amount: 4 DWORDs
  g1[1] = (int)((K & 0xffffu) << 16);                  // tensor_dim0[15:0]
  g1[2] = (int)(((K >> 16) & 0xffffu)                  // tensor_dim0[31:16]
        |       ((rows & 0xffffu) << 16));             // tensor_dim1[15:0]
  g1[3] = (int)(((rows >> 16) & 0xffffu)               // tensor_dim1[31:16]
        |       ((tile_k & 0xffffu) << 16));           // tile_dim0
  g1[4] = (int)(tile_rows & 0xffffu);                  // tile_dim1 (tile_dim2=0)
  g1[5] = (int)K;                                      // tensor_dim0_stride[31:0]
  g1[6] = 0;                                           // stride0[47:32], stride1 lo
  g1[7] = 0;
  i32x4 z4 = {};
#if __clang_major__ >= 23
  i32x8 z8 = {};
  __builtin_amdgcn_tensor_load_to_lds(g0, g1, z4, z4, z8, 0);
#else
  __builtin_amdgcn_tensor_load_to_lds(g0, g1, z4, z4, 0);
#endif
}
#else
#define HAVE_TDM 0
#endif

// ---------------------------------------------------------------------------
// fp32 -> bf16 conversion (grid-stride)
// ---------------------------------------------------------------------------
__global__ void k_cvt(const float* __restrict__ in, bf16* __restrict__ out, size_t n) {
  size_t i = (size_t)blockIdx.x * blockDim.x + threadIdx.x;
  size_t stride = (size_t)gridDim.x * blockDim.x;
  for (; i < n; i += stride) out[i] = f2bf(in[i]);
}

// ---------------------------------------------------------------------------
// Generic WMMA bf16 GEMM:  C[M,N] = A[M,K] * B[N,K]^T   (both row-major, K-contig)
// Block tile 128x128x32, 256 threads = 8 waves (2x4), wave tile 64x32.
// TDM double-buffered staging when available; fused epilogues.
// ---------------------------------------------------------------------------
enum { EP_F32 = 0, EP_BF16 = 1, EP_SILU_MUL = 2, EP_AXPY_BF16 = 3, EP_COMBINE = 4 };

#define GBM 128
#define GBK 32
#define LDS_STRIDE 40   // 32 + 8 pad elems (80B rows; TDM pad reproduces this)

// one K-step of WMMA work from a staged (A,B) tile pair
__device__ __forceinline__ void gemm_tile_step(
    const bf16* __restrict__ sAbuf, const bf16* __restrict__ sBbuf,
    int wrow, int wcol, int mlane, int klo, int kb16, v8f (&acc)[4][2]) {
  v16bf af[4], bfr[2];
#pragma unroll
  for (int mi = 0; mi < 4; mi++) {
    const bf16* base = &sAbuf[(wrow * 64 + mi * 16 + mlane) * LDS_STRIDE];
    af[mi] = ld16(base + klo, base + 16 + klo);
  }
#pragma unroll
  for (int ni = 0; ni < 2; ni++) {
    const bf16* base = &sBbuf[(wcol * 32 + ni * 16 + mlane) * LDS_STRIDE + klo];
    bfr[ni] = ld16(base, base + 16);
  }
#pragma unroll
  for (int mi = 0; mi < 4; mi++)
#pragma unroll
    for (int ni = 0; ni < 2; ni++)
      acc[mi][ni] = __builtin_amdgcn_wmma_f32_16x16x32_bf16(
          false, af[mi], false, bfr[ni], (short)0, acc[mi][ni], false, false);
}

template<int EP>
__global__ __launch_bounds__(256, 2)
void k_gemm(const bf16* __restrict__ A, const bf16* __restrict__ Bw,
            float* __restrict__ Cf, bf16* __restrict__ Cb,
            const float* __restrict__ aux, const unsigned char* __restrict__ hasv,
            int M, int N, int K)
{
  __shared__ bf16 sA[2][GBM * LDS_STRIDE];
  __shared__ bf16 sB[2][GBM * LDS_STRIDE];

  const int tid  = threadIdx.x;
  const int lane = tid & 31;
  const int wave = tid >> 5;
  const int wrow = wave >> 2;       // 0..1  -> 64 rows each
  const int wcol = wave & 3;        // 0..3  -> 32 cols each
  const int m0 = blockIdx.y * GBM;
  const int n0 = blockIdx.x * GBM;

  const int mlane = lane & 15;
  const int klo   = (lane >> 4) * 8;
  const int kb16  = (lane >> 4) * 16;

  v8f acc[4][2];
#pragma unroll
  for (int i = 0; i < 4; i++)
#pragma unroll
    for (int j = 0; j < 2; j++) acc[i][j] = vzero8();

#if HAVE_TDM
  // ---- TDM double-buffered pipeline: tile k+1 streams while tile k computes.
  const unsigned ldsA[2] = { (unsigned)(size_t)(void*)&sA[0][0],
                             (unsigned)(size_t)(void*)&sA[1][0] };
  const unsigned ldsB[2] = { (unsigned)(size_t)(void*)&sB[0][0],
                             (unsigned)(size_t)(void*)&sB[1][0] };
  const int nk = K / GBK;
  if (tid == 0) {
    tdm_load_2d(ldsA[0], A  + (size_t)m0 * K, GBK, GBM, (unsigned)K, (unsigned)M);
    tdm_load_2d(ldsB[0], Bw + (size_t)n0 * K, GBK, GBM, (unsigned)K, (unsigned)N);
    __builtin_amdgcn_s_wait_tensorcnt(0);
  }
  __syncthreads();
  for (int kt = 0; kt < nk; kt++) {
    const int buf = kt & 1;
    if (tid == 0 && kt + 1 < nk) {
      const size_t k1 = (size_t)(kt + 1) * GBK;
      tdm_load_2d(ldsA[buf ^ 1], A  + (size_t)m0 * K + k1, GBK, GBM,
                  (unsigned)K, (unsigned)M);
      tdm_load_2d(ldsB[buf ^ 1], Bw + (size_t)n0 * K + k1, GBK, GBM,
                  (unsigned)K, (unsigned)N);
    }
    gemm_tile_step(sA[buf], sB[buf], wrow, wcol, mlane, klo, kb16, acc);
    if (tid == 0 && kt + 1 < nk) __builtin_amdgcn_s_wait_tensorcnt(0);
    __syncthreads();
  }
#else
  // ---- fallback: cooperative synchronous staging + prefetch
  const int lrow  = tid >> 1;
  const int lhalf = (tid & 1) * 16;
  for (int k0 = 0; k0 < K; k0 += GBK) {
    {
      const bf16* ga = A  + (size_t)(m0 + lrow) * K + k0 + lhalf;
      const bf16* gb = Bw + (size_t)(n0 + lrow) * K + k0 + lhalf;
      *(uint4*)&sA[0][lrow * LDS_STRIDE + lhalf]     = *(const uint4*)ga;
      *(uint4*)&sA[0][lrow * LDS_STRIDE + lhalf + 8] = *(const uint4*)(ga + 8);
      *(uint4*)&sB[0][lrow * LDS_STRIDE + lhalf]     = *(const uint4*)gb;
      *(uint4*)&sB[0][lrow * LDS_STRIDE + lhalf + 8] = *(const uint4*)(gb + 8);
      if (k0 + GBK < K) {
        __builtin_prefetch(ga + GBK, 0, 0);
        __builtin_prefetch(gb + GBK, 0, 0);
      }
    }
    __syncthreads();
    gemm_tile_step(sA[0], sB[0], wrow, wcol, mlane, klo, kb16, acc);
    __syncthreads();
  }
#endif

  // -------- fused epilogue (C layout: VGPR r -> M = 8*(lane>=16)+r, N = lane&15)
  const int rbase = (lane >> 4) * 8;
#pragma unroll
  for (int mi = 0; mi < 4; mi++) {
#pragma unroll
    for (int ni = 0; ni < 2; ni++) {
#pragma unroll
      for (int r = 0; r < 8; r++) {
        const int row = m0 + wrow * 64 + mi * 16 + rbase + r;
        const int col = n0 + wcol * 32 + ni * 16 + mlane;
        const size_t idx = (size_t)row * N + col;
        const float c = acc[mi][ni][r];
        if constexpr (EP == EP_F32) {
          Cf[idx] = c;
        } else if constexpr (EP == EP_BF16) {
          Cb[idx] = f2bf(c);
        } else if constexpr (EP == EP_SILU_MUL) {       // hidden = silu(gate)*up
          const float s = c / (1.0f + __expf(-c));
          Cb[idx] = f2bf(s * aux[idx]);
        } else if constexpr (EP == EP_AXPY_BF16) {      // hidden += 0.1*adapt_proj
          const float v = bf2f(Cb[idx]) + 0.1f * c;
          Cb[idx] = f2bf(v);
        } else {                                        // out = shared + 0.1*mask*tok
          const float addv = hasv[row] ? 0.1f * c : 0.0f;
          Cf[idx] = aux[idx] + addv;
        }
      }
    }
  }
}

// ---------------------------------------------------------------------------
// LayerNorm over A=128 (one block per token), writes bf16
// ---------------------------------------------------------------------------
__global__ __launch_bounds__(128)
void k_ln128(const float* __restrict__ in, const float* __restrict__ g,
             const float* __restrict__ bta, bf16* __restrict__ outb)
{
  __shared__ float r1[128], r2[128];
  const int tok = blockIdx.x, c = threadIdx.x;
  const float v = in[(size_t)tok * 128 + c];
  r1[c] = v; r2[c] = v * v;
  __syncthreads();
  for (int s = 64; s > 0; s >>= 1) {
    if (c < s) { r1[c] += r1[c + s]; r2[c] += r2[c + s]; }
    __syncthreads();
  }
  const float mean = r1[0] * (1.0f / 128.0f);
  const float var  = r2[0] * (1.0f / 128.0f) - mean * mean;
  const float y = (v - mean) * __frsqrt_rn(var + 1e-5f) * g[c] + bta[c];
  outb[(size_t)tok * 128 + c] = f2bf(y);
}

// ---------------------------------------------------------------------------
// Fused S x S pseudo-attention (flash style, never materializes [B,S,S]).
// grid = (S/128, B). Per 64-col tile: P = silu(clip(Q*K^T)), adapt += P*V.
// V staged in natural [t][a] layout with b128 stores; the transpose happens on
// the (much smaller) fragment-gather side.
// ---------------------------------------------------------------------------
#define SVS 136   // V tile row stride: 128 + 8 pad (272B rows, 16B aligned)

__global__ __launch_bounds__(256)
void k_attn(const bf16* __restrict__ Qn, const bf16* __restrict__ Kn,
            bf16* __restrict__ outA, int S)
{
  __shared__ bf16 sP[128 * 72];   // P tile [128 rows][64 t + pad]
  __shared__ bf16 sV[64 * SVS];   // V tile [64 t][128 a + pad]

  const int b  = blockIdx.y;
  const int rt = blockIdx.x;
  const int tid = threadIdx.x, lane = tid & 31, wave = tid >> 5;
  const int wrow = wave >> 2, wcol = wave & 3;
  const size_t base = (size_t)b * S * 128;
  const bf16* Q  = Qn + base;
  const bf16* Kd = Kn + base;

  const int mlane = lane & 15;
  const int klo   = (lane >> 4) * 8;
  const int kb16  = (lane >> 4) * 16;
  const int rbase = (lane >> 4) * 8;

  v8f acc2[4][2];
#pragma unroll
  for (int i = 0; i < 4; i++)
#pragma unroll
    for (int j = 0; j < 2; j++) acc2[i][j] = vzero8();

  const int nct = S / 64;
  for (int ct = 0; ct < nct; ct++) {
    // stage V tile naturally: sV[t][a] (vectorized, coalesced)
    {
      const int tt   = tid >> 2;          // 0..63
      const int aseg = (tid & 3) * 32;    // 0,32,64,96
      const bf16* src = Q + (size_t)(ct * 64 + tt) * 128 + aseg;
      uint4* dst = (uint4*)&sV[tt * SVS + aseg];
#pragma unroll
      for (int j = 0; j < 4; j++) dst[j] = ((const uint4*)src)[j];
    }

    // ---- GEMM1: P(128x64) = Q(128x128) * K^T, wave tile 64x16, K loop 4x32
    v8f c1[4];
#pragma unroll
    for (int mi = 0; mi < 4; mi++) c1[mi] = vzero8();
    {
      const int tcol = ct * 64 + wcol * 16 + mlane;
      const bf16* kp = Kd + (size_t)tcol * 128;
#pragma unroll
      for (int ks = 0; ks < 4; ks++) {
        const v16bf bf_ = ld16(kp + ks * 32 + klo, kp + ks * 32 + 16 + klo);
#pragma unroll
        for (int mi = 0; mi < 4; mi++) {
          const int row = rt * 128 + wrow * 64 + mi * 16 + mlane;
          const bf16* qp = Q + (size_t)row * 128 + ks * 32;
          const v16bf qf = ld16(qp + klo, qp + 16 + klo);
          c1[mi] = __builtin_amdgcn_wmma_f32_16x16x32_bf16(
              false, qf, false, bf_, (short)0, c1[mi], false, false);
        }
      }
    }

    // silu(clip(.)) -> bf16 P tile in LDS
#pragma unroll
    for (int mi = 0; mi < 4; mi++) {
#pragma unroll
      for (int r = 0; r < 8; r++) {
        const int prow = wrow * 64 + mi * 16 + rbase + r;
        const int pcol = wcol * 16 + mlane;
        sP[prow * 72 + pcol] = f2bf(silu_clip(c1[mi][r]));
      }
    }
    __syncthreads();

    // ---- GEMM2: adapt(128x128) += P(128x64) * V(64x128), wave tile 64x32
#pragma unroll
    for (int ks = 0; ks < 2; ks++) {
      v16bf pf[4];
#pragma unroll
      for (int mi = 0; mi < 4; mi++) {
        const bf16* p = &sP[(wrow * 64 + mi * 16 + mlane) * 72 + ks * 32];
        pf[mi] = ld16(p + klo, p + 16 + klo);
      }
#pragma unroll
      for (int ni = 0; ni < 2; ni++) {
        const int acol = wcol * 32 + ni * 16 + mlane;
        v16bf vf;
#pragma unroll
        for (int e = 0; e < 16; e++)         // strided gather (transpose on read)
          vf[e] = sV[(ks * 32 + klo + e + (e >= 8 ? 8 : 0)) * SVS + acol];
#pragma unroll
        for (int mi = 0; mi < 4; mi++)
          acc2[mi][ni] = __builtin_amdgcn_wmma_f32_16x16x32_bf16(
              false, pf[mi], false, vf, (short)0, acc2[mi][ni], false, false);
      }
    }
    __syncthreads();
  }

  // write adapt (bf16)
#pragma unroll
  for (int mi = 0; mi < 4; mi++)
#pragma unroll
    for (int ni = 0; ni < 2; ni++)
#pragma unroll
      for (int r = 0; r < 8; r++) {
        const int row = rt * 128 + wrow * 64 + mi * 16 + rbase + r;
        const int col = wcol * 32 + ni * 16 + mlane;
        outA[base + (size_t)row * 128 + col] = f2bf(acc2[mi][ni][r]);
      }
}

// ---------------------------------------------------------------------------
// Per-token expert: pick LAST positive expert, sel = LN(w_ea[e] @ h), flag has.
// ---------------------------------------------------------------------------
__global__ __launch_bounds__(128)
void k_expert(const float* __restrict__ hpre, const float* __restrict__ ew,
              const float* __restrict__ w_ea, const float* __restrict__ elng,
              const float* __restrict__ elnb, bf16* __restrict__ selb,
              unsigned char* __restrict__ hasv)
{
  __shared__ float hsh[128];
  __shared__ float r1[128], r2[128];
  const int tok = blockIdx.x, c = threadIdx.x;
  hsh[c] = hpre[(size_t)tok * 128 + c];

  int e = -1;                               // uniform across block
#pragma unroll
  for (int i = 0; i < 8; i++)
    if (ew[(size_t)tok * 8 + i] > 0.0f) e = i;
  __syncthreads();

  if (e < 0) {                              // uniform branch
    selb[(size_t)tok * 128 + c] = f2bf(0.0f);
    if (c == 0) hasv[tok] = 0;
    return;
  }

  const float* wr = w_ea + ((size_t)e * 128 + c) * 128;
  float acc = 0.0f;
#pragma unroll 4
  for (int a = 0; a < 128; a++) acc += wr[a] * hsh[a];

  r1[c] = acc; r2[c] = acc * acc;
  __syncthreads();
  for (int s = 64; s > 0; s >>= 1) {
    if (c < s) { r1[c] += r1[c + s]; r2[c] += r2[c + s]; }
    __syncthreads();
  }
  const float mean = r1[0] * (1.0f / 128.0f);
  const float var  = r2[0] * (1.0f / 128.0f) - mean * mean;
  const float y = (acc - mean) * __frsqrt_rn(var + 1e-5f)
                  * elng[e * 128 + c] + elnb[e * 128 + c];
  selb[(size_t)tok * 128 + c] = f2bf(y);
  if (c == 0) hasv[tok] = 1;
}

// ---------------------------------------------------------------------------
// Host launcher
// ---------------------------------------------------------------------------
extern "C" void kernel_launch(void* const* d_in, const int* in_sizes, int n_in,
                              void* d_out, int out_size, void* d_ws, size_t ws_size,
                              hipStream_t stream) {
  (void)in_sizes; (void)n_in; (void)out_size; (void)ws_size;
  const float* x      = (const float*)d_in[0];
  const float* ew     = (const float*)d_in[1];
  const float* w_up   = (const float*)d_in[2];
  const float* w_gate = (const float*)d_in[3];
  const float* w_down = (const float*)d_in[4];
  const float* w_pre  = (const float*)d_in[5];
  const float* w_post = (const float*)d_in[6];
  const float* ln_g   = (const float*)d_in[7];
  const float* ln_b   = (const float*)d_in[8];
  const float* w_apj  = (const float*)d_in[9];
  const float* w_ea   = (const float*)d_in[10];
  const float* eln_g  = (const float*)d_in[11];
  const float* eln_b  = (const float*)d_in[12];
  const float* w_ep   = (const float*)d_in[13];
  const float* w_op   = (const float*)d_in[14];
  float* out = (float*)d_out;

  char* ws = (char*)d_ws;
  size_t off = 0;
  auto alloc = [&](size_t bytes) -> char* {
    char* p = ws + off; off += (bytes + 255) & ~(size_t)255; return p;
  };
  bf16* xb     = (bf16*)alloc((size_t)Mc * Dc * 2);
  bf16* wupb   = (bf16*)alloc((size_t)Hc * Dc * 2);
  bf16* wgtb   = (bf16*)alloc((size_t)Hc * Dc * 2);
  bf16* wdnb   = (bf16*)alloc((size_t)Dc * Hc * 2);
  bf16* wpreb  = (bf16*)alloc((size_t)Ac * Dc * 2);
  bf16* wpostb = (bf16*)alloc((size_t)Ac * Hc * 2);
  bf16* wapjb  = (bf16*)alloc((size_t)Hc * Ac * 2);
  bf16* wepb   = (bf16*)alloc((size_t)Hc * Ac * 2);
  bf16* wopb   = (bf16*)alloc((size_t)Dc * Hc * 2);
  float* upf   = (float*)alloc((size_t)Mc * Hc * 4);   // recycled below
  bf16*  hidb  = (bf16*) alloc((size_t)Mc * Hc * 2);
  float* pref  = (float*)alloc((size_t)Mc * Ac * 4);
  float* postf = (float*)alloc((size_t)Mc * Ac * 4);
  bf16*  ainb  = (bf16*) alloc((size_t)Mc * Ac * 2);
  bf16*  aoutb = (bf16*) alloc((size_t)Mc * Ac * 2);
  bf16*  adpb  = (bf16*) alloc((size_t)Mc * Ac * 2);
  bf16*  selb  = (bf16*) alloc((size_t)Mc * Ac * 2);
  unsigned char* hasb = (unsigned char*)alloc((size_t)Mc);
  float* sharedf = upf;                                 // recycle up_f32 region
  bf16*  tok1b   = (bf16*)((char*)upf + (size_t)Mc * Dc * 4);

  const dim3 blk256(256);
  auto cvt = [&](const float* src, bf16* dst, size_t n) {
    k_cvt<<<dim3(2048), blk256, 0, stream>>>(src, dst, n);
  };
  cvt(x,      xb,     (size_t)Mc * Dc);
  cvt(w_up,   wupb,   (size_t)Hc * Dc);
  cvt(w_gate, wgtb,   (size_t)Hc * Dc);
  cvt(w_down, wdnb,   (size_t)Dc * Hc);
  cvt(w_pre,  wpreb,  (size_t)Ac * Dc);
  cvt(w_post, wpostb, (size_t)Ac * Hc);
  cvt(w_apj,  wapjb,  (size_t)Hc * Ac);
  cvt(w_ep,   wepb,   (size_t)Hc * Ac);
  cvt(w_op,   wopb,   (size_t)Dc * Hc);

  auto ggrid = [](int N, int M) { return dim3((unsigned)(N / 128), (unsigned)(M / 128)); };

  k_gemm<EP_F32><<<ggrid(Hc, Mc), blk256, 0, stream>>>(
      xb, wupb, upf, nullptr, nullptr, nullptr, Mc, Hc, Dc);
  k_gemm<EP_SILU_MUL><<<ggrid(Hc, Mc), blk256, 0, stream>>>(
      xb, wgtb, nullptr, hidb, upf, nullptr, Mc, Hc, Dc);
  k_gemm<EP_F32><<<ggrid(Ac, Mc), blk256, 0, stream>>>(
      xb, wpreb, pref, nullptr, nullptr, nullptr, Mc, Ac, Dc);
  k_ln128<<<dim3(Mc), dim3(128), 0, stream>>>(pref, ln_g, ln_b, ainb);
  k_gemm<EP_F32><<<ggrid(Ac, Mc), blk256, 0, stream>>>(
      hidb, wpostb, postf, nullptr, nullptr, nullptr, Mc, Ac, Hc);
  k_ln128<<<dim3(Mc), dim3(128), 0, stream>>>(postf, ln_g, ln_b, aoutb);
  k_attn<<<dim3(Sc / 128, Bc), blk256, 0, stream>>>(ainb, aoutb, adpb, Sc);
  k_gemm<EP_AXPY_BF16><<<ggrid(Hc, Mc), blk256, 0, stream>>>(
      adpb, wapjb, nullptr, hidb, nullptr, nullptr, Mc, Hc, Ac);
  k_gemm<EP_F32><<<ggrid(Dc, Mc), blk256, 0, stream>>>(
      hidb, wdnb, sharedf, nullptr, nullptr, nullptr, Mc, Dc, Hc);
  k_expert<<<dim3(Mc), dim3(128), 0, stream>>>(
      pref, ew, w_ea, eln_g, eln_b, selb, hasb);
  k_gemm<EP_BF16><<<ggrid(Hc, Mc), blk256, 0, stream>>>(
      selb, wepb, nullptr, tok1b, nullptr, nullptr, Mc, Hc, Ac);
  k_gemm<EP_COMBINE><<<ggrid(Dc, Mc), blk256, 0, stream>>>(
      tok1b, wopb, out, nullptr, sharedf, hasb, Mc, Dc, Hc);
}
